// Attention_47510928229112
// MI455X (gfx1250) — compile-verified, matrix operand order checked by reading, not run on hardware
//
#include <hip/hip_runtime.h>
#include <hip/hip_bf16.h>

#define BATCH 64
#define SEQ   4096
#define EDIM  512
#define DDIM  512
#define ADIM  256
#define NEGV  -1e10f

typedef __attribute__((ext_vector_type(16))) __bf16 v16bf;
typedef __attribute__((ext_vector_type(8)))  float  v8f;

union BF16x16 {
    v16bf v;
    unsigned int u[8];
};

__device__ __forceinline__ unsigned int pack_bf16(float lo, float hi) {
    unsigned int ul = __float_as_uint(lo);
    unsigned int uh = __float_as_uint(hi);
    ul = (ul + 0x7FFFu + ((ul >> 16) & 1u)) >> 16;   // round-to-nearest-even
    uh = (uh + 0x7FFFu + ((uh >> 16) & 1u)) >> 16;
    return (ul & 0xFFFFu) | (uh << 16);
}

// ---------------------------------------------------------------------------
// Stage 0a: W_enc [E][A] fp32  ->  W_encT [A][E] bf16 (bits in ushort)
// ---------------------------------------------------------------------------
__global__ __launch_bounds__(256) void prep_wenc_kernel(const float* __restrict__ W_enc,
                                                        unsigned short* __restrict__ wencT) {
    int idx = blockIdx.x * 256 + threadIdx.x;          // over A*E = 131072
    int a = idx / EDIM;
    int e = idx % EDIM;
    float x = W_enc[e * ADIM + a];
    unsigned int u = __float_as_uint(x);
    u = (u + 0x7FFFu + ((u >> 16) & 1u)) >> 16;
    wencT[idx] = (unsigned short)u;
}

// ---------------------------------------------------------------------------
// Stage 0b: dec_proj[b][a] = dec[b]·W_dec[:,a] + b_dec[a] + b_enc[a]
// ---------------------------------------------------------------------------
__global__ __launch_bounds__(256) void prep_dec_kernel(const float* __restrict__ dec,
                                                       const float* __restrict__ W_dec,
                                                       const float* __restrict__ b_dec,
                                                       const float* __restrict__ b_enc,
                                                       float* __restrict__ dec_proj) {
    int b = blockIdx.x;
    int a = threadIdx.x;                                // 256 threads == ADIM
    float acc = b_dec[a] + b_enc[a];
    const float* dh = dec + b * DDIM;
    #pragma unroll 8
    for (int d = 0; d < DDIM; ++d)
        acc += dh[d] * W_dec[d * ADIM + a];
    dec_proj[b * ADIM + a] = acc;
}

// ---------------------------------------------------------------------------
// Stage 1: scores[b][s] = W_a · tanh(enc[b,s]·W_enc + dec_proj[b]) + b_a  (masked)
// WMMA bf16: 8 waves/WG, wave owns 16 rows of S. W_encT staged in LDS in two
// 128-row (N) halves of 128 KB each.
// ---------------------------------------------------------------------------
__global__ __launch_bounds__(256) void scores_kernel(const float* __restrict__ enc,
                                                     const int* __restrict__ mask,
                                                     const unsigned short* __restrict__ wencT,
                                                     const float* __restrict__ dec_proj,
                                                     const float* __restrict__ W_a,
                                                     const float* __restrict__ b_a_p,
                                                     float* __restrict__ scores) {
    extern __shared__ unsigned short lw[];              // 128 rows x 512 K bf16 = 128 KB

    const int b    = blockIdx.y;
    const int wave = threadIdx.x >> 5;
    const int lane = threadIdx.x & 31;
    const int ln   = lane & 15;                          // M (A-op) / N (B-op) within tile
    const int hi   = lane >> 4;                          // K-half selector
    const int s0   = blockIdx.x * 128 + wave * 16;

    // ---- Preload this lane's A-operand row (M = ln) for all 16 K-steps ----
    // ISA A layout (16-bit 16x32): lanes 0-15 hold K 0..7 & 16..23 of row M,
    // lanes 16-31 hold K 8..15 & 24..31; K pairs packed per dword.
    BF16x16 areg[16];
    const float* rowp = enc + ((size_t)(b * SEQ + s0 + ln)) * EDIM;
    #pragma unroll
    for (int kt = 0; kt < 16; ++kt) {
        int k0 = kt * 32 + hi * 8;
        float4 f0 = *(const float4*)(rowp + k0);
        float4 f1 = *(const float4*)(rowp + k0 + 4);
        float4 f2 = *(const float4*)(rowp + k0 + 16);
        float4 f3 = *(const float4*)(rowp + k0 + 20);
        areg[kt].u[0] = pack_bf16(f0.x, f0.y);
        areg[kt].u[1] = pack_bf16(f0.z, f0.w);
        areg[kt].u[2] = pack_bf16(f1.x, f1.y);
        areg[kt].u[3] = pack_bf16(f1.z, f1.w);
        areg[kt].u[4] = pack_bf16(f2.x, f2.y);
        areg[kt].u[5] = pack_bf16(f2.z, f2.w);
        areg[kt].u[6] = pack_bf16(f3.x, f3.y);
        areg[kt].u[7] = pack_bf16(f3.z, f3.w);
    }

    float sacc[8];
    #pragma unroll
    for (int v = 0; v < 8; ++v) sacc[v] = 0.0f;

    for (int half = 0; half < 2; ++half) {
        // ---- Stage 128 N-rows of W_encT (bf16, K-contiguous) into LDS ----
        __syncthreads();
        {
            const uint4* src = (const uint4*)(wencT + half * 128 * EDIM);
            uint4* dst = (uint4*)lw;                     // 8192 uint4
            for (int i = threadIdx.x; i < 8192; i += 256)
                dst[i] = src[i];
        }
        __syncthreads();

        #pragma unroll
        for (int ntl = 0; ntl < 8; ++ntl) {
            const int n = half * 128 + ntl * 16 + ln;    // this lane's N column
            v8f c = {};
            const unsigned short* brow = lw + (ntl * 16 + ln) * EDIM;
            #pragma unroll
            for (int kt = 0; kt < 16; ++kt) {
                // B-operand (32x16, column-major per lane): lane=N, K-halves split at lane 16
                const unsigned short* bp = brow + kt * 32 + hi * 8;
                BF16x16 bv;
                uint4 q0 = *(const uint4*)(bp);
                uint4 q1 = *(const uint4*)(bp + 16);
                bv.u[0] = q0.x; bv.u[1] = q0.y; bv.u[2] = q0.z; bv.u[3] = q0.w;
                bv.u[4] = q1.x; bv.u[5] = q1.y; bv.u[6] = q1.z; bv.u[7] = q1.w;
                c = __builtin_amdgcn_wmma_f32_16x16x32_bf16(
                        false, areg[kt].v, false, bv.v, (short)0, c, false, false);
            }
            // C layout: VGPR v -> M = v (lanes 0-15) / v+8 (lanes 16-31); lane%16 = N
            const float dp = dec_proj[b * ADIM + n];
            const float wa = W_a[n];
            #pragma unroll
            for (int v = 0; v < 8; ++v)
                sacc[v] += wa * tanhf(c[v] + dp);
        }
    }

    // ---- Reduce over the 16 lanes of each N-group (stays within halves) ----
    #pragma unroll
    for (int off = 1; off < 16; off <<= 1) {
        #pragma unroll
        for (int v = 0; v < 8; ++v)
            sacc[v] += __shfl_xor(sacc[v], off, 32);
    }

    if (ln == 0) {                                      // lanes 0 (rows 0-7) and 16 (rows 8-15)
        const float ba = *b_a_p;
        const int mbase = hi * 8;
        #pragma unroll
        for (int v = 0; v < 8; ++v) {
            int s = s0 + mbase + v;
            float val = sacc[v] + ba;
            scores[b * SEQ + s] = (mask[b * SEQ + s] == 0) ? NEGV : val;
        }
    }
}

// ---------------------------------------------------------------------------
// Stage 2: softmax over S per batch; also zero the context output region
// ---------------------------------------------------------------------------
__global__ __launch_bounds__(256) void softmax_kernel(const float* __restrict__ scores,
                                                      float* __restrict__ attn,
                                                      float* __restrict__ ctx) {
    __shared__ float red[256];
    const int b = blockIdx.x;
    const int tid = threadIdx.x;
    const float* srow = scores + b * SEQ;
    float* arow = attn + b * SEQ;

    float m = -3.4e38f;
    for (int s = tid; s < SEQ; s += 256) m = fmaxf(m, srow[s]);
    red[tid] = m; __syncthreads();
    for (int o = 128; o > 0; o >>= 1) {
        if (tid < o) red[tid] = fmaxf(red[tid], red[tid + o]);
        __syncthreads();
    }
    m = red[0]; __syncthreads();

    float sum = 0.0f;
    for (int s = tid; s < SEQ; s += 256) {
        float e = __expf(srow[s] - m);
        arow[s] = e;
        sum += e;
    }
    red[tid] = sum; __syncthreads();
    for (int o = 128; o > 0; o >>= 1) {
        if (tid < o) red[tid] += red[tid + o];
        __syncthreads();
    }
    const float inv = 1.0f / red[0];

    for (int s = tid; s < SEQ; s += 256) arow[s] *= inv;

    // zero context region for the atomic accumulation pass
    for (int e = tid; e < EDIM; e += 256) ctx[b * EDIM + e] = 0.0f;
}

// ---------------------------------------------------------------------------
// Stage 3: context[b][e] += sum_s attn[b][s] * enc[b][s][e]  (512-row S chunks)
// ---------------------------------------------------------------------------
__global__ __launch_bounds__(256) void context_kernel(const float* __restrict__ enc,
                                                      const float* __restrict__ attn,
                                                      float* __restrict__ ctx) {
    __shared__ float att[512];
    const int b = blockIdx.y;
    const int schunk = blockIdx.x * 512;
    const int tid = threadIdx.x;

    for (int i = tid; i < 512; i += 256)
        att[i] = attn[b * SEQ + schunk + i];
    __syncthreads();

    const int e0 = tid * 2;
    const float* base = enc + ((size_t)(b * SEQ + schunk)) * EDIM + e0;
    float acc0 = 0.0f, acc1 = 0.0f;
    #pragma unroll 4
    for (int s = 0; s < 512; ++s) {
        float a = att[s];
        float2 v = *(const float2*)(base + (size_t)s * EDIM);
        acc0 += a * v.x;
        acc1 += a * v.y;
    }
    unsafeAtomicAdd(&ctx[b * EDIM + e0],     acc0);
    unsafeAtomicAdd(&ctx[b * EDIM + e0 + 1], acc1);
}

// ---------------------------------------------------------------------------
extern "C" void kernel_launch(void* const* d_in, const int* in_sizes, int n_in,
                              void* d_out, int out_size, void* d_ws, size_t ws_size,
                              hipStream_t stream) {
    const float* enc   = (const float*)d_in[0];   // [B,S,E]
    const float* dec   = (const float*)d_in[1];   // [B,D]
    const int*   mask  = (const int*)  d_in[2];   // [B,S]
    const float* W_enc = (const float*)d_in[3];   // [E,A]
    const float* b_enc = (const float*)d_in[4];   // [A]
    const float* W_dec = (const float*)d_in[5];   // [D,A]
    const float* b_dec = (const float*)d_in[6];   // [A]
    const float* W_a   = (const float*)d_in[7];   // [A]
    const float* b_a   = (const float*)d_in[8];   // scalar

    float* attn_out = (float*)d_out;                       // [B,S]
    float* ctx_out  = attn_out + BATCH * SEQ;              // [B,E]

    // workspace layout
    char* ws = (char*)d_ws;
    unsigned short* wencT    = (unsigned short*)ws;                     // 256 KB
    float*          dec_proj = (float*)(ws + 262144);                   // 64 KB
    float*          scores   = (float*)(ws + 262144 + 65536);           // 1 MB

    prep_wenc_kernel<<<(ADIM * EDIM) / 256, 256, 0, stream>>>(W_enc, wencT);
    prep_dec_kernel<<<BATCH, 256, 0, stream>>>(dec, W_dec, b_dec, b_enc, dec_proj);

    dim3 sgrid(SEQ / 128, BATCH);
    scores_kernel<<<sgrid, 256, 128 * 1024, stream>>>(enc, mask, wencT, dec_proj,
                                                      W_a, b_a, scores);

    softmax_kernel<<<BATCH, 256, 0, stream>>>(scores, attn_out, ctx_out);

    dim3 cgrid(SEQ / 512, BATCH);
    context_kernel<<<cgrid, 256, 0, stream>>>(enc, attn_out, ctx_out);
}
